// MOERec_20607253086259
// MI455X (gfx1250) — matrix operand order repaired and checked by reading, not compiled, not run on hardware
//
#include <hip/hip_runtime.h>
#include <hip/hip_bf16.h>
#include <float.h>

typedef __attribute__((ext_vector_type(16))) __bf16 bf16x16;
typedef __attribute__((ext_vector_type(8)))  float  floatx8;

__device__ __forceinline__ unsigned short f32_to_bf16(float f) {
  unsigned u = __builtin_bit_cast(unsigned, f);
  u += 0x7FFFu + ((u >> 16) & 1u);          // round-to-nearest-even
  return (unsigned short)(u >> 16);
}
__device__ __forceinline__ float bf16_to_f32(unsigned short s) {
  unsigned u = ((unsigned)s) << 16;
  return __builtin_bit_cast(float, u);
}
__device__ __forceinline__ float softplusf(float x) {
  return fmaxf(x, 0.0f) + log1pf(expf(-fabsf(x)));
}
__device__ __forceinline__ float norm_cdf(float z) {
  return 0.5f * (1.0f + erff(z * 0.70710678118654752440f)); // z/sqrt(2)
}

// ---------------------------------------------------------------------------
// Zero the 32 global accumulator bins (importance[16], load[16]).
__global__ void moe_zero(float* __restrict__ gbins) {
  if (threadIdx.x < 32) gbins[threadIdx.x] = 0.0f;
}

// ---------------------------------------------------------------------------
// Repack w_gate / w_noise (4096x16 f32 each) into the per-lane bf16 B-fragment
// layout used here with V_WMMA_F32_16X16X32_BF16 (contiguous K, same order as A):
//   fragment index g = t*4096 + chunk*32 + lane  (t: 0=gate 1=noise, chunk: K/32)
//   lane holds N = lane&15, K = chunk*32 + (lane>>4)*16 + j, j=0..15 contiguous.
// Each fragment = 16 bf16 = 32 bytes, so the main kernel loads B with 2x b128.
__global__ void moe_prepack(const float* __restrict__ wg,
                            const float* __restrict__ wn,
                            unsigned short* __restrict__ outB) {
  int g = blockIdx.x * 256 + threadIdx.x;   // 0 .. 8191
  if (g >= 8192) return;
  int t     = g >> 12;
  int rem   = g & 4095;
  int chunk = rem >> 5;
  int lane  = rem & 31;
  int n     = lane & 15;
  int kstart = chunk * 32 + ((lane >> 4) << 4);
  const float* src = t ? wn : wg;
  union { unsigned short s[16]; uint4 q[2]; } tmp;
#pragma unroll
  for (int j = 0; j < 16; ++j)
    tmp.s[j] = f32_to_bf16(src[(size_t)(kstart + j) * 16 + n]);
  uint4* dst = (uint4*)(outB + (size_t)g * 16);
  dst[0] = tmp.q[0];
  dst[1] = tmp.q[1];
}

// ---------------------------------------------------------------------------
// Fused MoE kernel: one block = 16 batch rows.
//  LDS: X tile bf16 (16x4096, 128KB) | prelu(a) f32 (16x128) | logits (2x16x16)
//       | bins (32) | per-row routing info.
__global__ __launch_bounds__(512) void moe_main(
    const float* __restrict__ muti, const float* __restrict__ amat,
    const float* __restrict__ prelu_w, const float* __restrict__ noise,
    const unsigned short* __restrict__ packedB,
    float* __restrict__ gbins, float* __restrict__ out, int B) {
  extern __shared__ char smem[];
  unsigned short* Xs   = (unsigned short*)smem;                   // 131072 B
  float* wp            = (float*)(smem + 131072);                 //   8192 B
  float* logits        = (float*)(smem + 131072 + 8192);          //   2048 B
  float* bins          = (float*)(smem + 131072 + 8192 + 2048);   //    128 B
  int*   ridx          = (int*)  (smem + 131072 + 8192 + 2048 + 128);       // 128 B
  float* rg            = (float*)(smem + 131072 + 8192 + 2048 + 128 + 128); // 128 B

  const int tid  = threadIdx.x;
  const int tile = blockIdx.x;
  const int b0   = tile * 16;

  // ---- Phase 0: init LDS, stream X tile (f32 -> bf16), build prelu weights.
  logits[tid & 511] = 0.0f;          // 512 threads cover 512 floats
  if (tid < 32) bins[tid] = 0.0f;

  {
    const float4* src = (const float4*)muti + (size_t)tile * 16384;
#pragma unroll
    for (int i = 0; i < 32; ++i) {
      int f = tid + i * 512;         // float4 index within tile (row = f>>10)
      int r = f >> 10;
      float4 v = make_float4(0.f, 0.f, 0.f, 0.f);
      if (b0 + r < B) v = src[f];
      unsigned lo = (unsigned)f32_to_bf16(v.x) | ((unsigned)f32_to_bf16(v.y) << 16);
      unsigned hi = (unsigned)f32_to_bf16(v.z) | ((unsigned)f32_to_bf16(v.w) << 16);
      *(uint2*)(Xs + (size_t)f * 4) = make_uint2(lo, hi);
    }
  }
  for (int i = tid; i < 2048; i += 512) {
    float av = amat[i];
    wp[i] = (av >= 0.0f) ? av : prelu_w[i >> 7] * av;
  }
  __syncthreads();

  // ---- Phase 1: gating GEMMs via bf16 WMMA (fp32 accumulate).
  // wave = t(0:gate,1:noise) x 8 K-slices of 512; 16 chained 16x16x32 WMMAs.
  {
    const int wave  = tid >> 5;
    const int lane  = tid & 31;
    const int t     = wave & 1;
    const int slice = wave >> 1;
    const int m     = lane & 15;     // A row / C column
    const int kh    = lane >> 4;     // K-half select
    floatx8 acc = {0.f, 0.f, 0.f, 0.f, 0.f, 0.f, 0.f, 0.f};
    const uint4* Bbase = (const uint4*)(packedB + (size_t)t * (128 * 32 * 16));
#pragma unroll 4
    for (int cc = 0; cc < 16; ++cc) {
      int chunk = slice * 16 + cc;
      int kb2   = chunk * 32 + kh * 16;
      // A (same K order as packed B): lane holds row m, K = kb2..kb2+15 contiguous.
      const uint4* xa = (const uint4*)(Xs + (size_t)m * 4096 + kb2);
      union { bf16x16 v; uint4 q[2]; } A, Bf;
      A.q[0] = xa[0];
      A.q[1] = xa[1];                // +8 bf16 elements = +16 bytes
      const uint4* bp = Bbase + (size_t)(chunk * 32 + lane) * 2;
      Bf.q[0] = bp[0];
      Bf.q[1] = bp[1];
      acc = __builtin_amdgcn_wmma_f32_16x16x32_bf16(
          false, A.v, false, Bf.v, (short)0, acc, false, false);
    }
    // C layout: VGPR g -> M = kh*8+g, N = m. Reduce 8 K-slice partials in LDS.
#pragma unroll
    for (int g = 0; g < 8; ++g)
      atomicAdd(&logits[t * 256 + (kh * 8 + g) * 16 + m], acc[g]);
  }
  __syncthreads();

  // ---- Phase 2: per-row routing (16 lanes, one per batch row).
  if (tid < 16) {
    int b = b0 + tid;
    if (b < B) {
      float clean[16], nz[16], sd[16];
#pragma unroll
      for (int n = 0; n < 16; ++n) {
        clean[n] = logits[tid * 16 + n];
        sd[n]    = softplusf(logits[256 + tid * 16 + n]) + 0.01f;
        nz[n]    = clean[n] + noise[(size_t)b * 16 + n] * sd[n];
      }
      float v1 = -FLT_MAX, v2 = -FLT_MAX, v3 = -FLT_MAX;
      int i1 = 0, i2 = 0;
#pragma unroll
      for (int n = 0; n < 16; ++n) {
        float x = nz[n];
        if (x > v1)      { v3 = v2; v2 = v1; i2 = i1; v1 = x; i1 = n; }
        else if (x > v2) { v3 = v2; v2 = x;  i2 = n; }
        else if (x > v3) { v3 = x; }
      }
      float e2  = expf(v2 - v1);
      float inv = 1.0f / (1.0f + e2);
      ridx[tid * 2]     = i1;
      ridx[tid * 2 + 1] = i2;
      rg[tid * 2]       = inv;        // softmax over {v1, v2}
      rg[tid * 2 + 1]   = e2 * inv;
      atomicAdd(&bins[i1], inv);      // importance
      atomicAdd(&bins[i2], e2 * inv);
#pragma unroll
      for (int n = 0; n < 16; ++n) {  // load statistic
        float thr = (nz[n] > v3) ? v3 : v2;   // thr_in = 3rd, thr_out = 2nd
        atomicAdd(&bins[16 + n], norm_cdf((clean[n] - thr) / sd[n]));
      }
    } else {
      ridx[tid * 2] = 0; ridx[tid * 2 + 1] = 0;
      rg[tid * 2] = 0.f; rg[tid * 2 + 1] = 0.f;
    }
  }
  __syncthreads();

  // ---- Phase 3: top-2 expert combine, reusing the LDS X tile.
  {
    int r = tid >> 5, c = tid & 31;
    int   n1 = ridx[r * 2], n2 = ridx[r * 2 + 1];
    float g1 = rg[r * 2],   g2 = rg[r * 2 + 1];
    const float* w1 = wp + n1 * 128;
    const float* w2 = wp + n2 * 128;
    const unsigned short* xr = Xs + (size_t)r * 4096 + c;
    float a1 = 0.f, a2 = 0.f;
#pragma unroll 8
    for (int e = 0; e < 128; ++e) {
      float xm = bf16_to_f32(xr[e * 32]);
      a1 = fmaf(xm, w1[e], a1);
      a2 = fmaf(xm, w2[e], a2);
    }
    int b = b0 + r;
    if (b < B) {
      float comb = g1 * expf(a1) + g2 * expf(a2);
      if (comb == 0.0f) comb = 2.2204460492503131e-16f;  // float32(np.finfo(float).eps)
      out[(size_t)b * 32 + c] = logf(comb);
    }
  }
  __syncthreads();
  if (tid < 32) atomicAdd(&gbins[tid], bins[tid]);  // 32 atomics per block
}

// ---------------------------------------------------------------------------
// loss = cv2(importance) + cv2(load), cv2(x) = var(x, ddof=1)/(mean^2 + 1e-10)
__global__ void moe_loss(const float* __restrict__ gbins,
                         float* __restrict__ out, int B) {
  if (blockIdx.x == 0 && threadIdx.x == 0) {
    float loss = 0.0f;
    for (int t = 0; t < 2; ++t) {
      const float* v = gbins + t * 16;
      float s = 0.0f;
      for (int i = 0; i < 16; ++i) s += v[i];
      float mean = s * (1.0f / 16.0f);
      float var = 0.0f;
      for (int i = 0; i < 16; ++i) { float d = v[i] - mean; var += d * d; }
      var *= (1.0f / 15.0f);
      loss += var / (mean * mean + 1e-10f);
    }
    out[(size_t)B * 32] = loss;
  }
}

// ---------------------------------------------------------------------------
extern "C" void kernel_launch(void* const* d_in, const int* in_sizes, int n_in,
                              void* d_out, int out_size, void* d_ws, size_t ws_size,
                              hipStream_t stream) {
  const float* muti    = (const float*)d_in[0];
  const float* w_gate  = (const float*)d_in[1];
  const float* w_noise = (const float*)d_in[2];
  const float* amat    = (const float*)d_in[3];
  const float* prelu_w = (const float*)d_in[4];
  const float* noise   = (const float*)d_in[5];
  float* out = (float*)d_out;

  const int B = in_sizes[0] / 4096;            // muti_int is (B,128,32)
  char* ws = (char*)d_ws;
  unsigned short* packedB = (unsigned short*)ws;      // 262144 bytes
  float* gbins = (float*)(ws + 262144);               // importance[16], load[16]

  constexpr size_t SMEM = 131072 + 8192 + 2048 + 128 + 128 + 128; // 141696 B
  (void)hipFuncSetAttribute((const void*)moe_main,
                            hipFuncAttributeMaxDynamicSharedMemorySize,
                            (int)SMEM);

  moe_zero<<<1, 32, 0, stream>>>(gbins);
  moe_prepack<<<32, 256, 0, stream>>>(w_gate, w_noise, packedB);
  const int tiles = (B + 15) / 16;
  moe_main<<<tiles, 512, SMEM, stream>>>(muti, amat, prelu_w, noise,
                                         packedB, gbins, out, B);
  moe_loss<<<1, 32, 0, stream>>>(gbins, out, B);
}
